// Multimodal_TransformerEncoder_68745246539828
// MI455X (gfx1250) — compile-verified, matrix operand order checked by reading, not run on hardware
//
#include <hip/hip_runtime.h>
#include <hip/hip_bf16.h>
#include <math.h>

// ---------------------------------------------------------------------------
// Types for CDNA5 WMMA (wave32): 16x16x32 bf16 -> f32
// ---------------------------------------------------------------------------
typedef __attribute__((ext_vector_type(16))) __bf16 v16bf;
typedef __attribute__((ext_vector_type(8)))  __bf16 v8bf;
typedef __attribute__((ext_vector_type(8)))  float  v8f;
typedef int v4i_g __attribute__((vector_size(16)));   // GNU vector int4

union Frag16 { v16bf v; v8bf h[2]; };

#if defined(__has_builtin)
#if __has_builtin(__builtin_amdgcn_global_load_async_to_lds_b128)
#define HAVE_ASYNC_LDS 1
#endif
#endif

__device__ __forceinline__ __bf16 f2bf(float f) {
  union { float f; unsigned u; } in; in.f = f;
  unsigned r = in.u + 0x7FFFu + ((in.u >> 16) & 1u);   // round-to-nearest-even
  union { unsigned short s; __bf16 b; } out;
  out.s = (unsigned short)(r >> 16);
  return out.b;
}

__device__ __forceinline__ v8f wmma_bf16(v16bf a, v16bf b, v8f c) {
  return __builtin_amdgcn_wmma_f32_16x16x32_bf16(false, a, false, b, (short)0, c,
                                                 false, false);
}

// Load one 16x32 fragment (A layout; also used for B supplied as B^T rows).
// Per lane (wave32): row = lane%16, halves [0..7] at k = 8*(lane/16),
// halves [8..15] at k = 16 + 8*(lane/16). stride in bf16 elements.
__device__ __forceinline__ v16bf load_frag(const __bf16* base, int stride) {
  int lane = threadIdx.x & 31;
  int r    = lane & 15;
  int kh   = (lane >> 4) * 8;
  const __bf16* p = base + (size_t)r * stride;
  Frag16 f;
  f.h[0] = *(const v8bf*)(p + kh);
  f.h[1] = *(const v8bf*)(p + 16 + kh);
  return f.v;
}

// Async 16B global -> LDS copy (ASYNCcnt path) with synchronous fallback.
// Probe-derived signature: (AS1 v4i*, AS3 v4i*, imm offset, imm cpol).
__device__ __forceinline__ void async_copy16(__bf16* ldsDst, const __bf16* gsrc) {
#if defined(HAVE_ASYNC_LDS)
  __builtin_amdgcn_global_load_async_to_lds_b128(
      (__attribute__((address_space(1))) v4i_g*)gsrc,
      (__attribute__((address_space(3))) v4i_g*)ldsDst, 0, 0);
#else
  *(v8bf*)ldsDst = *(const v8bf*)gsrc;
#endif
}

__device__ __forceinline__ void async_wait0() {
#if defined(HAVE_ASYNC_LDS)
#if __has_builtin(__builtin_amdgcn_s_wait_asynccnt)
  __builtin_amdgcn_s_wait_asynccnt(0);
#else
  asm volatile("s_wait_asynccnt 0" ::: "memory");
#endif
#endif
}

// ---------------------------------------------------------------------------
// fp32 -> bf16 elementwise convert
// ---------------------------------------------------------------------------
__global__ __launch_bounds__(256) void cvt_kernel(const float* __restrict__ x,
                                                  __bf16* __restrict__ y, int n) {
  int i = blockIdx.x * 256 + threadIdx.x;
  if (i < n) y[i] = f2bf(x[i]);
}

// ---------------------------------------------------------------------------
// Weight transpose + convert: w[K,N] fp32 -> wT[N,K] bf16 (LDS 32x32 tiles)
// ---------------------------------------------------------------------------
__global__ __launch_bounds__(256) void tconv_kernel(const float* __restrict__ w,
                                                    __bf16* __restrict__ wT,
                                                    int K, int N) {
  __shared__ float tile[32][33];
  int tx = threadIdx.x, ty = threadIdx.y;           // (32, 8)
  int kb = blockIdx.y * 32, nb = blockIdx.x * 32;
#pragma unroll
  for (int r = 0; r < 4; ++r)
    tile[ty * 4 + r][tx] = w[(size_t)(kb + ty * 4 + r) * N + nb + tx];
  __syncthreads();
#pragma unroll
  for (int r = 0; r < 4; ++r)
    wT[(size_t)(nb + ty * 4 + r) * K + kb + tx] = f2bf(tile[tx][ty * 4 + r]);
}

// ---------------------------------------------------------------------------
// WMMA GEMM: C[M,N] = A[M,K](bf16) * BT[N,K](bf16)^T + bias, optional leaky.
// Block = 8 waves; wave computes a 32x64 tile (2 A frags x 4 B frags = 8
// WMMAs per 32-deep K step). Ping-pong double buffer unrolled x64 in K:
// each WMMA burst consumes one fragment stage while the other stage's loads
// are in flight -- no register rotation copies, no WMMA/VALU hazard NOPs.
// Requires K % 64 == 0 (K is 1024 or 4096 here).
// ---------------------------------------------------------------------------
__global__ __launch_bounds__(256) void gemm_bf16_kernel(
    const __bf16* __restrict__ A, const __bf16* __restrict__ BT,
    const float* __restrict__ bias, float* __restrict__ Cf,
    __bf16* __restrict__ Cb, int M, int N, int K, int act) {
  int wave = threadIdx.x >> 5;
  int lane = threadIdx.x & 31;
  int m0 = blockIdx.y * 256 + wave * 32;
  int n0 = blockIdx.x * 64;
  const __bf16* arow0 = A + (size_t)m0 * K;
  const __bf16* arow1 = A + (size_t)(m0 + 16) * K;
  const __bf16* brow[4];
#pragma unroll
  for (int nt = 0; nt < 4; ++nt) brow[nt] = BT + (size_t)(n0 + nt * 16) * K;

  v8f acc[2][4] = {};

  // stage A <- k = 0
  v16bf a0A = load_frag(arow0, K);
  v16bf a1A = load_frag(arow1, K);
  v16bf bA[4];
#pragma unroll
  for (int nt = 0; nt < 4; ++nt) bA[nt] = load_frag(brow[nt], K);

  v16bf a0B, a1B, bB[4];
  int k = 0;
  for (; k + 64 < K; k += 64) {
    // stage B <- k+32 (issues while stage-A WMMAs run)
    a0B = load_frag(arow0 + k + 32, K);
    a1B = load_frag(arow1 + k + 32, K);
#pragma unroll
    for (int nt = 0; nt < 4; ++nt) bB[nt] = load_frag(brow[nt] + k + 32, K);
#pragma unroll
    for (int nt = 0; nt < 4; ++nt) {
      acc[0][nt] = wmma_bf16(a0A, bA[nt], acc[0][nt]);
      acc[1][nt] = wmma_bf16(a1A, bA[nt], acc[1][nt]);
    }
    // stage A <- k+64 (issues while stage-B WMMAs run)
    a0A = load_frag(arow0 + k + 64, K);
    a1A = load_frag(arow1 + k + 64, K);
#pragma unroll
    for (int nt = 0; nt < 4; ++nt) bA[nt] = load_frag(brow[nt] + k + 64, K);
#pragma unroll
    for (int nt = 0; nt < 4; ++nt) {
      acc[0][nt] = wmma_bf16(a0B, bB[nt], acc[0][nt]);
      acc[1][nt] = wmma_bf16(a1B, bB[nt], acc[1][nt]);
    }
  }
  // tail: k = K-64 (stage A already holds it), then K-32
  a0B = load_frag(arow0 + k + 32, K);
  a1B = load_frag(arow1 + k + 32, K);
#pragma unroll
  for (int nt = 0; nt < 4; ++nt) bB[nt] = load_frag(brow[nt] + k + 32, K);
#pragma unroll
  for (int nt = 0; nt < 4; ++nt) {
    acc[0][nt] = wmma_bf16(a0A, bA[nt], acc[0][nt]);
    acc[1][nt] = wmma_bf16(a1A, bA[nt], acc[1][nt]);
  }
#pragma unroll
  for (int nt = 0; nt < 4; ++nt) {
    acc[0][nt] = wmma_bf16(a0B, bB[nt], acc[0][nt]);
    acc[1][nt] = wmma_bf16(a1B, bB[nt], acc[1][nt]);
  }

  int lhi = lane >> 4, ln = lane & 15;
#pragma unroll
  for (int half = 0; half < 2; ++half)
#pragma unroll
    for (int nt = 0; nt < 4; ++nt) {
      int col = n0 + nt * 16 + ln;
      float bv = bias ? bias[col] : 0.f;
#pragma unroll
      for (int i = 0; i < 8; ++i) {
        int row = m0 + half * 16 + i + 8 * lhi;
        float v = acc[half][nt][i] + bv;
        if (act) v = v > 0.f ? v : 0.01f * v;
        size_t idx = (size_t)row * N + col;
        if (Cf) Cf[idx] = v;
        if (Cb) Cb[idx] = f2bf(v);
      }
    }
}

// ---------------------------------------------------------------------------
// Per-head L2 normalize: t[65536,64] f32 -> out bf16 (wave per row)
// ---------------------------------------------------------------------------
__global__ __launch_bounds__(256) void l2n_kernel(const float* __restrict__ t,
                                                  __bf16* __restrict__ out) {
  int wave = threadIdx.x >> 5, lane = threadIdx.x & 31;
  size_t row = (size_t)blockIdx.x * 8 + wave;
  const float* p = t + row * 64;
  float a = p[lane], b = p[lane + 32];
  float ss = a * a + b * b;
#pragma unroll
  for (int d = 1; d < 32; d <<= 1) ss += __shfl_xor(ss, d, 32);
  float inv = 1.f / fmaxf(sqrtf(ss), 1e-12f);
  out[row * 64 + lane]      = f2bf(a * inv);
  out[row * 64 + lane + 32] = f2bf(b * inv);
}

// ---------------------------------------------------------------------------
// Flash-style attention. Block = (b,h,128 q rows); wave owns 16 q rows.
// scores = (qn @ kn^T masked)/8 ; online softmax ; out = P @ V ; bf16 out.
// K-tile staged via GLOBAL_LOAD_ASYNC_TO_LDS_B128 when available.
// ---------------------------------------------------------------------------
__global__ __launch_bounds__(256) void attn_kernel(
    const __bf16* __restrict__ qn, const __bf16* __restrict__ kn,
    const __bf16* __restrict__ vb, const int* __restrict__ mask,
    __bf16* __restrict__ out) {
  __shared__ __attribute__((aligned(16))) __bf16 sk[32 * 64];   // 32 keys x 64 d
  __shared__ __attribute__((aligned(16))) __bf16 svt[64 * 32];  // d-major V tile
  __shared__ __attribute__((aligned(16))) __bf16 sp[8][16 * 32];// per-wave P
  int wave = threadIdx.x >> 5, lane = threadIdx.x & 31;
  int lhi = lane >> 4, ln = lane & 15;
  int qblk = blockIdx.x & 3;
  int h    = (blockIdx.x >> 2) & 15;
  int bidx = blockIdx.x >> 6;
  int q0   = qblk * 128 + wave * 16;                 // q row within batch

  const __bf16* qbase = qn + ((size_t)(bidx * 512 + q0) * 1024) + h * 64;
  v16bf qf0 = load_frag(qbase, 1024);
  v16bf qf1 = load_frag(qbase + 32, 1024);

  float m[8], l[8];
#pragma unroll
  for (int i = 0; i < 8; ++i) { m[i] = -INFINITY; l[i] = 0.f; }
  v8f o[4] = {};

  for (int kb = 0; kb < 512; kb += 32) {
    __syncthreads();
    {  // cooperative staging: K tile row-major (async), V tile transposed
      int key  = threadIdx.x >> 3;
      int doff = (threadIdx.x & 7) * 8;
      size_t src = ((size_t)(bidx * 512 + kb + key) * 1024) + h * 64 + doff;
      async_copy16(&sk[key * 64 + doff], kn + src);
      v8bf vv = *(const v8bf*)(vb + src);
#pragma unroll
      for (int j = 0; j < 8; ++j) svt[(doff + j) * 32 + key] = vv[j];
    }
    async_wait0();
    __syncthreads();

    float s[2][8];
#pragma unroll
    for (int ks = 0; ks < 2; ++ks) {
      v16bf kf0 = load_frag(&sk[ks * 16 * 64], 64);
      v16bf kf1 = load_frag(&sk[ks * 16 * 64 + 32], 64);
      v8f c = {};
      c = wmma_bf16(qf0, kf0, c);
      c = wmma_bf16(qf1, kf1, c);
#pragma unroll
      for (int i = 0; i < 8; ++i) {
        int qg = q0 + i + 8 * lhi;
        int kg = kb + ks * 16 + ln;
        float e = c[i];
        if (mask[((size_t)bidx * 512 + qg) * 512 + kg] == 1) e = -1e-8f;
        s[ks][i] = e * 0.125f;  // 1/sqrt(64)
      }
    }

    float sc[8];
#pragma unroll
    for (int i = 0; i < 8; ++i) {
      float lm = fmaxf(s[0][i], s[1][i]);
#pragma unroll
      for (int d = 1; d < 16; d <<= 1) lm = fmaxf(lm, __shfl_xor(lm, d, 32));
      float mn = fmaxf(m[i], lm);
      sc[i] = __expf(m[i] - mn);
      float p0 = __expf(s[0][i] - mn);
      float p1 = __expf(s[1][i] - mn);
      float ps = p0 + p1;
#pragma unroll
      for (int d = 1; d < 16; d <<= 1) ps += __shfl_xor(ps, d, 32);
      l[i] = l[i] * sc[i] + ps;
      m[i] = mn;
      int row = i + 8 * lhi;
      sp[wave][row * 32 + ln]      = f2bf(p0);
      sp[wave][row * 32 + 16 + ln] = f2bf(p1);
    }
#pragma unroll
    for (int nt = 0; nt < 4; ++nt)
#pragma unroll
      for (int i = 0; i < 8; ++i) o[nt][i] = o[nt][i] * sc[i];

    v16bf pf = load_frag(&sp[wave][0], 32);
#pragma unroll
    for (int nt = 0; nt < 4; ++nt) {
      v16bf vf = load_frag(&svt[nt * 16 * 32], 32);
      o[nt] = wmma_bf16(pf, vf, o[nt]);
    }
  }

#pragma unroll
  for (int nt = 0; nt < 4; ++nt)
#pragma unroll
    for (int i = 0; i < 8; ++i) {
      size_t row = (size_t)bidx * 512 + q0 + i + 8 * lhi;
      int col = h * 64 + nt * 16 + ln;
      out[row * 1024 + col] = f2bf(o[nt][i] / l[i]);
    }
}

// ---------------------------------------------------------------------------
// Residual + LayerNorm: out = LN(y + res)*g + b ; writes f32 and bf16 copies.
// ---------------------------------------------------------------------------
__global__ __launch_bounds__(256) void ln_kernel(
    const float* __restrict__ y, const float* __restrict__ res,
    const float* __restrict__ g, const float* __restrict__ bb,
    float* __restrict__ outf, __bf16* __restrict__ outb) {
  int row = blockIdx.x, t = threadIdx.x;
  const float* yp = y + (size_t)row * 1024;
  const float* rp = res + (size_t)row * 1024;
  float v[4], s = 0.f, ss = 0.f;
#pragma unroll
  for (int j = 0; j < 4; ++j) {
    v[j] = yp[t * 4 + j] + rp[t * 4 + j];
    s += v[j]; ss += v[j] * v[j];
  }
#pragma unroll
  for (int d = 1; d < 32; d <<= 1) { s += __shfl_xor(s, d, 32); ss += __shfl_xor(ss, d, 32); }
  __shared__ float as[8], ass[8];
  int wave = t >> 5, lane = t & 31;
  if (lane == 0) { as[wave] = s; ass[wave] = ss; }
  __syncthreads();
  s = 0.f; ss = 0.f;
#pragma unroll
  for (int w = 0; w < 8; ++w) { s += as[w]; ss += ass[w]; }
  float mu  = s * (1.f / 1024.f);
  float var = ss * (1.f / 1024.f) - mu * mu;
  float inv = rsqrtf(var + 1e-5f);
#pragma unroll
  for (int j = 0; j < 4; ++j) {
    int col = t * 4 + j;
    float o = (v[j] - mu) * inv * g[col] + bb[col];
    outf[(size_t)row * 1024 + col] = o;
    outb[(size_t)row * 1024 + col] = f2bf(o);
  }
}

// ---------------------------------------------------------------------------
// Host launcher
// ---------------------------------------------------------------------------
extern "C" void kernel_launch(void* const* d_in, const int* in_sizes, int n_in,
                              void* d_out, int out_size, void* d_ws, size_t ws_size,
                              hipStream_t stream) {
  (void)in_sizes; (void)out_size; (void)ws_size;
  const float* query = (const float*)d_in[0];
  const float* value = (const float*)d_in[1];
  const int*   mask  = (const int*)d_in[n_in - 1];

  const size_t MB = 1024ull * 1024ull;
  char* w = (char*)d_ws;
  __bf16* xb   = (__bf16*)(w + 0 * MB);    // [4096,1024] bf16 activation in
  __bf16* valb = (__bf16*)(w + 8 * MB);    // bf16 value input
  __bf16* qnb  = (__bf16*)(w + 16 * MB);   // normalized q
  __bf16* knb  = (__bf16*)(w + 24 * MB);   // normalized k
  __bf16* vbb  = (__bf16*)(w + 32 * MB);   // v
  __bf16* aob  = (__bf16*)(w + 40 * MB);   // attention out
  __bf16* x1b  = (__bf16*)(w + 48 * MB);   // LN1 out bf16
  __bf16* f1b  = (__bf16*)(w + 56 * MB);   // [4096,4096] FFN hidden
  __bf16* wT   = (__bf16*)(w + 88 * MB);   // transposed bf16 weight (<=8MB)
  float*  t0   = (float*)(w + 96 * MB);    // f32 scratch
  float*  t1   = (float*)(w + 112 * MB);   // f32 scratch
  float*  X    = (float*)(w + 128 * MB);   // residual chain
  float*  X1   = (float*)(w + 144 * MB);   // post-LN1

  const int M = 4096, E = 1024, F = 4096;
  dim3 tb(32, 8);

  cvt_kernel<<<(M * E) / 256, 256, 0, stream>>>(query, xb, M * E);
  cvt_kernel<<<(M * E) / 256, 256, 0, stream>>>(value, valb, M * E);

  for (int lyr = 0; lyr < 4; ++lyr) {
    int base = 2 + lyr * 16;
    const float* wq = (const float*)d_in[base + 0];
    const float* bq = (const float*)d_in[base + 1];
    const float* wk = (const float*)d_in[base + 2];
    const float* bk = (const float*)d_in[base + 3];
    const float* wv = (const float*)d_in[base + 4];
    const float* bv = (const float*)d_in[base + 5];
    const float* wo = (const float*)d_in[base + 6];
    const float* bo = (const float*)d_in[base + 7];
    const float* w1 = (const float*)d_in[base + 8];
    const float* b1 = (const float*)d_in[base + 9];
    const float* w2 = (const float*)d_in[base + 10];
    const float* b2 = (const float*)d_in[base + 11];
    const float* g1 = (const float*)d_in[base + 12];
    const float* l1 = (const float*)d_in[base + 13];
    const float* g2 = (const float*)d_in[base + 14];
    const float* l2 = (const float*)d_in[base + 15];

    const float* resid = (lyr == 0) ? query : X;
    float* outx = (lyr == 3) ? (float*)d_out : X;

    // Q = x @ wq + bq -> l2norm -> qnb
    tconv_kernel<<<dim3(32, 32), tb, 0, stream>>>(wq, wT, E, E);
    gemm_bf16_kernel<<<dim3(E / 64, M / 256), 256, 0, stream>>>(xb, wT, bq, t0, nullptr, M, E, E, 0);
    l2n_kernel<<<8192, 256, 0, stream>>>(t0, qnb);
    // K = value @ wk + bk -> l2norm -> knb
    tconv_kernel<<<dim3(32, 32), tb, 0, stream>>>(wk, wT, E, E);
    gemm_bf16_kernel<<<dim3(E / 64, M / 256), 256, 0, stream>>>(valb, wT, bk, t0, nullptr, M, E, E, 0);
    l2n_kernel<<<8192, 256, 0, stream>>>(t0, knb);
    // V = value @ wv + bv (bf16)
    tconv_kernel<<<dim3(32, 32), tb, 0, stream>>>(wv, wT, E, E);
    gemm_bf16_kernel<<<dim3(E / 64, M / 256), 256, 0, stream>>>(valb, wT, bv, nullptr, vbb, M, E, E, 0);
    // attention
    attn_kernel<<<512, 256, 0, stream>>>(qnb, knb, vbb, mask, aob);
    // O projection
    tconv_kernel<<<dim3(32, 32), tb, 0, stream>>>(wo, wT, E, E);
    gemm_bf16_kernel<<<dim3(E / 64, M / 256), 256, 0, stream>>>(aob, wT, bo, t1, nullptr, M, E, E, 0);
    // LN1(o + x)
    ln_kernel<<<4096, 256, 0, stream>>>(t1, resid, g1, l1, X1, x1b);
    // FFN1 (leaky relu, bf16 out)
    tconv_kernel<<<dim3(F / 32, E / 32), tb, 0, stream>>>(w1, wT, E, F);
    gemm_bf16_kernel<<<dim3(F / 64, M / 256), 256, 0, stream>>>(x1b, wT, b1, nullptr, f1b, M, F, E, 1);
    // FFN2
    tconv_kernel<<<dim3(E / 32, F / 32), tb, 0, stream>>>(w2, wT, F, E);
    gemm_bf16_kernel<<<dim3(E / 64, M / 256), 256, 0, stream>>>(f1b, wT, b2, t0, nullptr, M, E, F, 0);
    // LN2(f2 + x1) -> next x (f32 + bf16)
    ln_kernel<<<4096, 256, 0, stream>>>(t0, X1, g2, l2, outx, xb);
  }
}
